// GraphConvolution_5669356833703
// MI455X (gfx1250) — matrix operand order repaired and checked by reading, not compiled, not run on hardware
//
#include <hip/hip_runtime.h>
#include <hip/hip_bf16.h>

// Types for CDNA5 WMMA (wave32): 16x16x32 f16 -> f32
typedef _Float16 v16h __attribute__((ext_vector_type(16)));
typedef _Float16 v8h  __attribute__((ext_vector_type(8)));
typedef _Float16 v4h  __attribute__((ext_vector_type(4)));
typedef float    v8f  __attribute__((ext_vector_type(8)));
typedef float    v4f  __attribute__((ext_vector_type(4)));
// Matches the builtin's parameter type: int __attribute__((vector_size(16)))
typedef int      gv4i __attribute__((vector_size(16)));

#define BATCH 8
#define NR    5
#define NN    1024
#define F_IN  128
#define F_H   128
#define F_OUT 64

// ---- CDNA5 async global->LDS path (guarded; fallback = load + ds_store) ----
#if defined(__gfx1250__) && __has_builtin(__builtin_amdgcn_global_load_async_to_lds_b128)
#define HAVE_ASYNC_LDS 1
#else
#define HAVE_ASYNC_LDS 0
#endif

#if HAVE_ASYNC_LDS
#if __has_builtin(__builtin_amdgcn_s_wait_asynccnt)
#define WAIT_ASYNC(n) __builtin_amdgcn_s_wait_asynccnt(n)
#else
#define WAIT_ASYNC(n) asm volatile("s_wait_asynccnt " #n ::: "memory")
#endif
#else
#define WAIT_ASYNC(n) ((void)0)
#endif

// LDS staging layout for a 16 x 128 f16 A-chunk: padded row stride of
// 136 f16 (= 272 B = 68 dwords; 68 mod 64 = 4 -> the 16 row starts hit
// distinct 4-dword-aligned banks, so 32-lane ds_load_b128 is conflict-free).
#define KCHUNK  128
#define LDS_ROW 136

// ---------------------------------------------------------------------------
// A-operand tile: 16x32 f16 at (m0,k0), row-major, leading dim ld.
// ISA layout: lanes 0-15: M=lane, K=kh*8+0..7 (v0-3) / 16+kh*8+0..7 (v4-7).
// ---------------------------------------------------------------------------
__device__ __forceinline__ v16h load_a_tile(const _Float16* __restrict__ src,
                                            int ld, int lane) {
  const int r  = lane & 15;
  const int kh = lane >> 4;
  const _Float16* p = src + (long long)r * ld + kh * 8;
  v16h a;
  *((v8h*)&a)     = *(const v8h*)(p);
  *((v8h*)&a + 1) = *(const v8h*)(p + 16);
  return a;
}

// Same tile sourced from the padded LDS chunk; s = k-subtile (0..3).
__device__ __forceinline__ v16h load_a_lds(const _Float16* As, int s, int lane) {
  const int r  = lane & 15;
  const int kh = lane >> 4;
  const _Float16* p = As + r * LDS_ROW + s * 32 + kh * 8;
  v16h a;
  *((v8h*)&a)     = *(const v8h*)(p);
  *((v8h*)&a + 1) = *(const v8h*)(p + 16);
  return a;
}

// ---------------------------------------------------------------------------
// B-operand tile: 32x16 f16 B(k,n), source stored transposed srcT[n][k].
// lanes 0-15: K=0..7,16..23 of col N=lane; lanes 16-31: K=8..15,24..31.
// ---------------------------------------------------------------------------
__device__ __forceinline__ v16h load_b_tile(const _Float16* __restrict__ srcT,
                                            int ldk, int lane) {
  const int c  = lane & 15;
  const int kh = lane >> 4;
  const _Float16* p = srcT + (long long)c * ldk + kh * 8;
  v16h b;
  *((v8h*)&b)     = *(const v8h*)(p);
  *((v8h*)&b + 1) = *(const v8h*)(p + 16);
  return b;
}

__device__ __forceinline__ v8f wmma_f16(v16h a, v16h b, v8f c) {
  return __builtin_amdgcn_wmma_f32_16x16x32_f16(
      false, a, false, b, (short)0, c, false, false);
}

// Stage 64 B of one A row into LDS (4 x b128 per staging lane).
__device__ __forceinline__ void stage_chunk(const _Float16* __restrict__ g,
                                            _Float16* __restrict__ l) {
#pragma unroll
  for (int j = 0; j < 4; ++j) {
#if HAVE_ASYNC_LDS
    __builtin_amdgcn_global_load_async_to_lds_b128(
        (__attribute__((address_space(1))) gv4i*)(g + j * 8),
        (__attribute__((address_space(3))) gv4i*)(l + j * 8), 0, 0);
#else
    *(v8h*)(l + j * 8) = *(const v8h*)(g + j * 8);
#endif
  }
}

// ---------------------------------------------------------------------------
// fp32 -> f16 conversion (weights / x)
// ---------------------------------------------------------------------------
__global__ void cvt_f32_to_f16(const float* __restrict__ in,
                               _Float16* __restrict__ out, int n) {
  int i = blockIdx.x * blockDim.x + threadIdx.x;
  if (i < n) out[i] = (_Float16)in[i];
}

// ---------------------------------------------------------------------------
// A = sum_r adj[:,r]. 160 MB streamed once with NT hint (keep L2 for A),
// result written f16 (16 MB, L2-resident for the two big GEMMs).
// grid: BATCH*1024 blocks x 256 thr, v4f per thread.
// ---------------------------------------------------------------------------
__global__ void reduce_adj(const float* __restrict__ adj,
                           _Float16* __restrict__ Abf) {
  const long long nn = (long long)NN * NN;
  const int b = blockIdx.x >> 10;                       // 1024 blocks per batch
  const long long rem = (((long long)(blockIdx.x & 1023)) * 256 + threadIdx.x) * 4;
  const float* p = adj + b * NR * nn + rem;
  v4f s = __builtin_nontemporal_load((const v4f*)p);
#pragma unroll
  for (int r = 1; r < NR; ++r)
    s += __builtin_nontemporal_load((const v4f*)(p + (long long)r * nn));
  v4h h;
  h[0] = (_Float16)s[0]; h[1] = (_Float16)s[1];
  h[2] = (_Float16)s[2]; h[3] = (_Float16)s[3];
  *(v4h*)(Abf + b * nn + rem) = h;
}

// ---------------------------------------------------------------------------
// h1 = x @ W1^T + b1 -> transposed h1T[b][feat][node] (f16). K = 128.
// ---------------------------------------------------------------------------
__global__ __launch_bounds__(256) void gemm_h1(
    const _Float16* __restrict__ xbf, const _Float16* __restrict__ W1bf,
    const float* __restrict__ b1, _Float16* __restrict__ h1T) {
  const int b = blockIdx.x >> 6;
  const int m0 = (blockIdx.x & 63) * 16;
  const int wave = threadIdx.x >> 5;
  const int lane = threadIdx.x & 31;
  const int n0 = wave * 16;
  const int col = lane & 15, row8 = (lane >> 4) * 8;

  const _Float16* Asrc = xbf + ((long long)b * NN + m0) * F_IN;
  const float bias = b1[n0 + col];
  v8f c;
#pragma unroll
  for (int g = 0; g < 8; ++g) c[g] = bias;

#pragma unroll
  for (int k0 = 0; k0 < F_IN; k0 += 32) {
    v16h a  = load_a_tile(Asrc + k0, F_IN, lane);
    v16h bm = load_b_tile(W1bf + (long long)n0 * F_IN + k0, F_IN, lane);
    c = wmma_f16(a, bm, c);
  }
  _Float16* dst = h1T + ((long long)b * F_H + n0 + col) * NN + m0 + row8;
#pragma unroll
  for (int g = 0; g < 8; ++g) dst[g] = (_Float16)c[g];
}

// ---------------------------------------------------------------------------
// hidden = A @ h1 + h1 -> row-major hid[b][node][feat] (f16). K = 1024.
// 8 waves; A-strip staged to LDS in 128-wide chunks (async, double-buffered).
// ---------------------------------------------------------------------------
__global__ __launch_bounds__(256) void gemm_hidden(
    const _Float16* __restrict__ Abf, const _Float16* __restrict__ h1T,
    _Float16* __restrict__ hid) {
  __shared__ __align__(16) _Float16 smA[2][16 * LDS_ROW];
  const int b = blockIdx.x >> 6;
  const int m0 = (blockIdx.x & 63) * 16;
  const int wave = threadIdx.x >> 5;
  const int lane = threadIdx.x & 31;
  const int n0 = wave * 16;
  const int col = lane & 15, row8 = (lane >> 4) * 8;

  const _Float16* Asrc = Abf + ((long long)b * NN + m0) * NN;
  const _Float16* Bsrc = h1T + ((long long)b * F_H + n0) * NN;
  const _Float16* skip = h1T + ((long long)b * F_H + n0 + col) * NN + m0 + row8;
  v8f c;
#pragma unroll
  for (int g = 0; g < 8; ++g) c[g] = (float)skip[g];   // + h1 skip

  // Staging lanes (wave 0): lane -> (row, 64B half-row)
  const int srow = lane >> 1, shalf = lane & 1;
  const _Float16* gstage = Asrc + (long long)srow * NN + shalf * 64;
  _Float16* lstage[2] = {&smA[0][srow * LDS_ROW + shalf * 64],
                         &smA[1][srow * LDS_ROW + shalf * 64]};
  if (wave == 0) stage_chunk(gstage, lstage[0]);       // prefetch chunk 0

  const int nchunks = NN / KCHUNK;                     // 8
  for (int ci = 0; ci < nchunks; ++ci) {
    const int cur = ci & 1;
    __syncthreads();                                   // prior reads of buf cur^1 done
    if (wave == 0) {
      const int knext = (ci + 1 < nchunks) ? (ci + 1) * KCHUNK : ci * KCHUNK;
      stage_chunk(gstage + knext, lstage[cur ^ 1]);
      WAIT_ASYNC(4);                                   // chunk ci landed
    }
    __syncthreads();                                   // publish buf cur
    const _Float16* As = &smA[cur][0];
#pragma unroll
    for (int s = 0; s < 4; ++s) {
      v16h a  = load_a_lds(As, s, lane);
      v16h bm = load_b_tile(Bsrc + ci * KCHUNK + s * 32, NN, lane);
      c = wmma_f16(a, bm, c);
    }
  }
  _Float16* dst = hid + ((long long)b * NN + m0) * F_H + n0 + col;
#pragma unroll
  for (int g = 0; g < 8; ++g)
    dst[(long long)(row8 + g) * F_H] = (_Float16)c[g];
}

// ---------------------------------------------------------------------------
// h2 = hidden @ W2^T + b2 -> transposed h2T[b][feat][node] (f16). K = 128.
// ---------------------------------------------------------------------------
__global__ __launch_bounds__(128) void gemm_h2(
    const _Float16* __restrict__ hid, const _Float16* __restrict__ W2bf,
    const float* __restrict__ b2, _Float16* __restrict__ h2T) {
  const int b = blockIdx.x >> 6;
  const int m0 = (blockIdx.x & 63) * 16;
  const int wave = threadIdx.x >> 5;
  const int lane = threadIdx.x & 31;
  const int n0 = wave * 16;
  const int col = lane & 15, row8 = (lane >> 4) * 8;

  const _Float16* Asrc = hid + ((long long)b * NN + m0) * F_H;
  const float bias = b2[n0 + col];
  v8f c;
#pragma unroll
  for (int g = 0; g < 8; ++g) c[g] = bias;

#pragma unroll
  for (int k0 = 0; k0 < F_H; k0 += 32) {
    v16h a  = load_a_tile(Asrc + k0, F_H, lane);
    v16h bm = load_b_tile(W2bf + (long long)n0 * F_H + k0, F_H, lane);
    c = wmma_f16(a, bm, c);
  }
  _Float16* dst = h2T + ((long long)b * F_OUT + n0 + col) * NN + m0 + row8;
#pragma unroll
  for (int g = 0; g < 8; ++g) dst[g] = (_Float16)c[g];
}

// ---------------------------------------------------------------------------
// out = A @ h2 + h2 -> fp32 [b][node][F_OUT]. K = 1024. 4 waves; LDS-staged A.
// ---------------------------------------------------------------------------
__global__ __launch_bounds__(128) void gemm_out(
    const _Float16* __restrict__ Abf, const _Float16* __restrict__ h2T,
    float* __restrict__ out) {
  __shared__ __align__(16) _Float16 smA[2][16 * LDS_ROW];
  const int b = blockIdx.x >> 6;
  const int m0 = (blockIdx.x & 63) * 16;
  const int wave = threadIdx.x >> 5;
  const int lane = threadIdx.x & 31;
  const int n0 = wave * 16;
  const int col = lane & 15, row8 = (lane >> 4) * 8;

  const _Float16* Asrc = Abf + ((long long)b * NN + m0) * NN;
  const _Float16* Bsrc = h2T + ((long long)b * F_OUT + n0) * NN;
  const _Float16* skip = h2T + ((long long)b * F_OUT + n0 + col) * NN + m0 + row8;
  v8f c;
#pragma unroll
  for (int g = 0; g < 8; ++g) c[g] = (float)skip[g];   // + h2 skip

  const int srow = lane >> 1, shalf = lane & 1;
  const _Float16* gstage = Asrc + (long long)srow * NN + shalf * 64;
  _Float16* lstage[2] = {&smA[0][srow * LDS_ROW + shalf * 64],
                         &smA[1][srow * LDS_ROW + shalf * 64]};
  if (wave == 0) stage_chunk(gstage, lstage[0]);

  const int nchunks = NN / KCHUNK;
  for (int ci = 0; ci < nchunks; ++ci) {
    const int cur = ci & 1;
    __syncthreads();
    if (wave == 0) {
      const int knext = (ci + 1 < nchunks) ? (ci + 1) * KCHUNK : ci * KCHUNK;
      stage_chunk(gstage + knext, lstage[cur ^ 1]);
      WAIT_ASYNC(4);
    }
    __syncthreads();
    const _Float16* As = &smA[cur][0];
#pragma unroll
    for (int s = 0; s < 4; ++s) {
      v16h a  = load_a_lds(As, s, lane);
      v16h bm = load_b_tile(Bsrc + ci * KCHUNK + s * 32, NN, lane);
      c = wmma_f16(a, bm, c);
    }
  }
  float* dst = out + ((long long)b * NN + m0) * F_OUT + n0 + col;
#pragma unroll
  for (int g = 0; g < 8; ++g)
    __builtin_nontemporal_store(c[g], dst + (long long)(row8 + g) * F_OUT);
}

// ---------------------------------------------------------------------------
// Launcher
// ---------------------------------------------------------------------------
extern "C" void kernel_launch(void* const* d_in, const int* in_sizes, int n_in,
                              void* d_out, int out_size, void* d_ws, size_t ws_size,
                              hipStream_t stream) {
  (void)in_sizes; (void)n_in; (void)out_size; (void)ws_size;

  const float* x   = (const float*)d_in[0];   // [8,1024,128]
  const float* adj = (const float*)d_in[1];   // [8,5,1024,1024]
  const float* W1  = (const float*)d_in[2];   // [128,128]
  const float* b1  = (const float*)d_in[3];   // [128]
  const float* W2  = (const float*)d_in[4];   // [64,128]
  const float* b2  = (const float*)d_in[5];   // [64]
  float* out       = (float*)d_out;           // [8,1024,64]

  char* ws = (char*)d_ws;
  _Float16* Abf   = (_Float16*)(ws + 0);                         // 16 MB
  _Float16* xbf   = (_Float16*)(ws + 16777216);                  //  2 MB
  _Float16* h1T   = (_Float16*)(ws + 18874368);                  //  2 MB
  _Float16* hid   = (_Float16*)(ws + 20971520);                  //  2 MB
  _Float16* h2T   = (_Float16*)(ws + 23068672);                  //  1 MB
  _Float16* W1bf  = (_Float16*)(ws + 24117248);                  // 32 KB
  _Float16* W2bf  = (_Float16*)(ws + 24150016);                  // 16 KB

  cvt_f32_to_f16<<<(BATCH * NN * F_IN + 255) / 256, 256, 0, stream>>>(x, xbf, BATCH * NN * F_IN);
  cvt_f32_to_f16<<<(F_H * F_IN + 255) / 256, 256, 0, stream>>>(W1, W1bf, F_H * F_IN);
  cvt_f32_to_f16<<<(F_OUT * F_H + 255) / 256, 256, 0, stream>>>(W2, W2bf, F_OUT * F_H);

  reduce_adj<<<BATCH * 1024, 256, 0, stream>>>(adj, Abf);

  gemm_h1<<<BATCH * (NN / 16), 256, 0, stream>>>(xbf, W1bf, b1, h1T);
  gemm_hidden<<<BATCH * (NN / 16), 256, 0, stream>>>(Abf, h1T, hid);
  gemm_h2<<<BATCH * (NN / 16), 128, 0, stream>>>(hid, W2bf, b2, h2T);
  gemm_out<<<BATCH * (NN / 16), 128, 0, stream>>>(Abf, h2T, out);
}
